// GRUEncoderDecoder_12893491822765
// MI455X (gfx1250) — matrix operand order repaired and checked by reading, not compiled, not run on hardware
//
#include <hip/hip_runtime.h>
#include <math.h>

// ---------------------------------------------------------------------------
// CDNA5 (gfx1250) wave32 WMMA types & helpers
// ---------------------------------------------------------------------------
typedef __attribute__((ext_vector_type(16))) _Float16 v16h;
typedef __attribute__((ext_vector_type(8)))  _Float16 v8h;
typedef __attribute__((ext_vector_type(8)))  float    v8f;

// A-matrix fragment (16x32 f16, MxK). Lane L: M = L&15, hi = L>>4.
// Per ISA layout: lane-half 0 VGPRs hold K {0..7, 16..23}, half 1 {8..15, 24..31}.
// => two contiguous 16B loads at half-element offsets 8*hi and 16+8*hi.
__device__ __forceinline__ v16h frag_A(const _Float16* __restrict__ rowk, int hi) {
  v8h a = *(const v8h*)(rowk + 8 * hi);
  v8h b = *(const v8h*)(rowk + 16 + 8 * hi);
  v16h r;
#pragma unroll
  for (int i = 0; i < 8; ++i) { r[i] = a[i]; r[i + 8] = b[i]; }
  return r;
}

// B-matrix fragment (32x16 f16, KxN) for B = W^T with W row-major (N x K):
// lane L: N = L&15 (a row of W), lane-half 0 holds K {0..7, 16..23}, half 1 {8..15, 24..31},
// matching frag_A => two contiguous 16B loads at 8*hi and 16+8*hi.
__device__ __forceinline__ v16h frag_B(const _Float16* __restrict__ rowk, int hi) {
  v8h a = *(const v8h*)(rowk + 8 * hi);
  v8h b = *(const v8h*)(rowk + 16 + 8 * hi);
  v16h r;
#pragma unroll
  for (int i = 0; i < 8; ++i) { r[i] = a[i]; r[i + 8] = b[i]; }
  return r;
}

__device__ __forceinline__ v8f wmma_f16(v16h a, v16h b, v8f c) {
  return __builtin_amdgcn_wmma_f32_16x16x32_f16(false, a, false, b, (short)0, c,
                                                false, false);
}

__device__ __forceinline__ float sigm(float x) { return 1.f / (1.f + __expf(-x)); }

// ---------------------------------------------------------------------------
// Fused GRU cell step:  for a 16-wide slice of hidden columns, compute all
// three gate rows over the full batch (B=64 -> 4 waves of 16-row M tiles),
// then apply gates and write h_new (and optionally y).
//   gates[b, g*H+j] = sum_k x[b,k] Wih[g*H+j,k] + sum_k h[b,k] Whh[g*H+j,k]
//   r=sig(.); z=sig(.); n=tanh(gi_n + r*gh_n); h'=(1-z)n+z h
// grid.x = H/16, block = 128 threads (4 waves)
// ---------------------------------------------------------------------------
__global__ __launch_bounds__(128) void gru_step_wmma(
    const _Float16* __restrict__ x, int ldx, int Kx,
    const _Float16* __restrict__ h_old, int ldh, int H,
    const _Float16* __restrict__ Wih,   // (3H x Kx) row-major, f16
    const _Float16* __restrict__ Whh,   // (3H x H)  row-major, f16
    const float* __restrict__ bih, const float* __restrict__ bhh,
    _Float16* __restrict__ h_new, int ldhn,
    _Float16* __restrict__ y_out, int ldy)
{
  const int lane = threadIdx.x & 31;
  const int wave = threadIdx.x >> 5;
  const int hi   = lane >> 4;
  const int lo   = lane & 15;
  const int j    = blockIdx.x * 16 + lo;  // hidden column
  const int m0   = wave * 16;             // batch row tile base

  v8f acc_r, acc_z, acc_ni, acc_nh;
  {
    float br = bih[j] + bhh[j];
    float bz = bih[H + j] + bhh[H + j];
    float bi = bih[2 * H + j];
    float bh = bhh[2 * H + j];
#pragma unroll
    for (int i = 0; i < 8; ++i) { acc_r[i] = br; acc_z[i] = bz; acc_ni[i] = bi; acc_nh[i] = bh; }
  }

  // ---- x @ Wih^T  (r, z, n-input parts) ----
  {
    const _Float16* arow = x + (size_t)(m0 + lo) * ldx;
    const _Float16* wr = Wih + (size_t)j * Kx;
    const _Float16* wz = Wih + (size_t)(H + j) * Kx;
    const _Float16* wn = Wih + (size_t)(2 * H + j) * Kx;
    for (int k = 0; k < Kx; k += 32) {
      v16h a  = frag_A(arow + k, hi);
      v16h b0 = frag_B(wr + k, hi);
      v16h b1 = frag_B(wz + k, hi);
      v16h b2 = frag_B(wn + k, hi);
      acc_r  = wmma_f16(a, b0, acc_r);
      acc_z  = wmma_f16(a, b1, acc_z);
      acc_ni = wmma_f16(a, b2, acc_ni);
    }
  }
  // ---- h @ Whh^T  (r, z, n-hidden parts) ----
  {
    const _Float16* arow = h_old + (size_t)(m0 + lo) * ldh;
    const _Float16* ur = Whh + (size_t)j * H;
    const _Float16* uz = Whh + (size_t)(H + j) * H;
    const _Float16* un = Whh + (size_t)(2 * H + j) * H;
    for (int k = 0; k < H; k += 32) {
      v16h a = frag_A(arow + k, hi);
      acc_r  = wmma_f16(a, frag_B(ur + k, hi), acc_r);
      acc_z  = wmma_f16(a, frag_B(uz + k, hi), acc_z);
      acc_nh = wmma_f16(a, frag_B(un + k, hi), acc_nh);
    }
  }

  // ---- gates (C layout: lane-half hi -> rows m0+8*hi+i, col j) ----
#pragma unroll
  for (int i = 0; i < 8; ++i) {
    const int row = m0 + 8 * hi + i;
    float r  = sigm(acc_r[i]);
    float z  = sigm(acc_z[i]);
    float n  = tanhf(acc_ni[i] + r * acc_nh[i]);
    float hp = (float)h_old[(size_t)row * ldh + j];
    float hv = (1.f - z) * n + z * hp;
    h_new[(size_t)row * ldhn + j] = (_Float16)hv;
    if (y_out) y_out[(size_t)row * ldy + j] = (_Float16)hv;
  }
}

// ---------------------------------------------------------------------------
// Generic NT GEMM: out[m,n] = sum_k A[m,k]*W[n,k] + bias[n], M=64 fixed.
// grid.x = N/16, block = 128 (4 waves x 16-row M tiles). f32 output.
// ---------------------------------------------------------------------------
__global__ __launch_bounds__(128) void gemm_nt_wmma(
    const _Float16* __restrict__ A, int lda,
    const _Float16* __restrict__ W, int ldw,
    const float* __restrict__ bias,
    float* __restrict__ out, int ldo, int K)
{
  const int lane = threadIdx.x & 31;
  const int wave = threadIdx.x >> 5;
  const int hi   = lane >> 4;
  const int lo   = lane & 15;
  const int n    = blockIdx.x * 16 + lo;
  const int m0   = wave * 16;

  v8f acc;
  const float bv = bias ? bias[n] : 0.f;
#pragma unroll
  for (int i = 0; i < 8; ++i) acc[i] = bv;

  const _Float16* arow = A + (size_t)(m0 + lo) * lda;
  const _Float16* wrow = W + (size_t)n * ldw;
  for (int k = 0; k < K; k += 32)
    acc = wmma_f16(frag_A(arow + k, hi), frag_B(wrow + k, hi), acc);

#pragma unroll
  for (int i = 0; i < 8; ++i)
    out[(size_t)(m0 + 8 * hi + i) * ldo + n] = acc[i];
}

// ---------------------------------------------------------------------------
// enc_score[b,s] = enc_out[b,s,:] . We + attn_b   (one wave per (b,s))
// y1 layout: (S, B, D)
// ---------------------------------------------------------------------------
__global__ __launch_bounds__(256) void enc_score_k(
    const _Float16* __restrict__ y1, const float* __restrict__ attnW,
    const float* __restrict__ attnB, float* __restrict__ encScore,
    int B, int S, int D)
{
  const int gw   = (int)((blockIdx.x * blockDim.x + threadIdx.x) >> 5);
  const int lane = threadIdx.x & 31;
  if (gw >= B * S) return;
  const int b = gw / S, s = gw % S;
  const _Float16* row = y1 + ((size_t)s * B + b) * (size_t)D;
  float acc = 0.f;
  for (int d = lane; d < D; d += 32) acc += (float)row[d] * attnW[D + d];
#pragma unroll
  for (int off = 16; off > 0; off >>= 1) acc += __shfl_down(acc, off);
  if (lane == 0) encScore[b * S + s] = acc + attnB[0];
}

// ---------------------------------------------------------------------------
// Attention + context + decoder input assembly. One block (256 thr) per batch.
// xbuf[b] = [ tgt_emb[tok] | softmax(enc_score + h1.Wq) @ enc_out ]
// Requires S == blockDim.x == 256.
// ---------------------------------------------------------------------------
__global__ __launch_bounds__(256) void attn_ctx_k(
    const float* __restrict__ encScore, const _Float16* __restrict__ h1, int ldh1,
    const float* __restrict__ attnW, const _Float16* __restrict__ y1,
    const int* __restrict__ tgt, int t, int T,
    const _Float16* __restrict__ tgtEmb, _Float16* __restrict__ xbuf,
    int B, int S, int D, int E)
{
  __shared__ float sm[256];
  __shared__ float pr[256];
  const int b = blockIdx.x, tid = threadIdx.x;

  float acc = 0.f;
  for (int d = tid; d < D; d += 256) acc += (float)h1[(size_t)b * ldh1 + d] * attnW[d];
  sm[tid] = acc; __syncthreads();
  for (int off = 128; off > 0; off >>= 1) { if (tid < off) sm[tid] += sm[tid + off]; __syncthreads(); }
  const float qdot = sm[0];
  __syncthreads();

  const float sc = encScore[b * S + tid] + qdot;
  sm[tid] = sc; __syncthreads();
  for (int off = 128; off > 0; off >>= 1) { if (tid < off) sm[tid] = fmaxf(sm[tid], sm[tid + off]); __syncthreads(); }
  const float mx = sm[0];
  __syncthreads();
  const float ex = __expf(sc - mx);
  sm[tid] = ex; __syncthreads();
  for (int off = 128; off > 0; off >>= 1) { if (tid < off) sm[tid] += sm[tid + off]; __syncthreads(); }
  const float inv = 1.f / sm[0];
  pr[tid] = ex * inv;
  __syncthreads();

  const int tok = tgt[b * T + t];
  for (int e = tid; e < E; e += 256)
    xbuf[(size_t)b * (E + D) + e] = tgtEmb[(size_t)tok * E + e];
  for (int d = tid; d < D; d += 256) {
    float c = 0.f;
    for (int s = 0; s < S; ++s)
      c += pr[s] * (float)y1[((size_t)s * B + b) * D + d];
    xbuf[(size_t)b * (E + D) + E + d] = (_Float16)c;
  }
}

// ---------------------------------------------------------------------------
// Small utility kernels
// ---------------------------------------------------------------------------
__global__ void f32_to_f16_k(const float* __restrict__ a, _Float16* __restrict__ b, size_t n) {
  size_t i = (size_t)blockIdx.x * blockDim.x + threadIdx.x;
  if (i < n) b[i] = (_Float16)a[i];
}
__global__ void zero_f16_k(_Float16* __restrict__ p, int n) {
  int i = blockIdx.x * blockDim.x + threadIdx.x;
  if (i < n) p[i] = (_Float16)0.f;
}
__global__ void zero_t0_k(float* __restrict__ out, int B, int T, int Vt) {
  int i = blockIdx.x * blockDim.x + threadIdx.x;
  if (i >= B * Vt) return;
  int b = i / Vt, v = i % Vt;
  out[(size_t)b * T * Vt + v] = 0.f;
}
__global__ void embed_src_k(const int* __restrict__ src, const float* __restrict__ emb,
                            _Float16* __restrict__ xs, int B, int S, int E, long total) {
  long i = (long)blockIdx.x * blockDim.x + threadIdx.x;
  if (i >= total) return;
  int e = (int)(i % E);
  long sb = i / E;
  int b = (int)(sb % B);
  int s = (int)(sb / B);
  xs[i] = (_Float16)emb[(size_t)src[b * S + s] * E + e];
}

// ---------------------------------------------------------------------------
// Host orchestration
// ---------------------------------------------------------------------------
extern "C" void kernel_launch(void* const* d_in, const int* in_sizes, int n_in,
                              void* d_out, int out_size, void* d_ws, size_t ws_size,
                              hipStream_t stream) {
  (void)in_sizes; (void)n_in; (void)out_size; (void)ws_size;
  const int B = 64, S = 256, T = 128, E = 256, H = 512, D = 1024, Vt = 8192;
  const int G3 = 3 * H;    // 1536
  const int DG3 = 3 * D;   // 3072

  const int*   src     = (const int*)  d_in[0];
  const int*   tgt     = (const int*)  d_in[1];
  const float* src_emb = (const float*)d_in[2];
  const float* tgt_emb = (const float*)d_in[3];
  const float* eWih0   = (const float*)d_in[4];
  const float* eWhh0   = (const float*)d_in[5];
  const float* ebih0   = (const float*)d_in[6];
  const float* ebhh0   = (const float*)d_in[7];
  const float* eWih1   = (const float*)d_in[8];
  const float* eWhh1   = (const float*)d_in[9];
  const float* ebih1   = (const float*)d_in[10];
  const float* ebhh1   = (const float*)d_in[11];
  const float* dWih0   = (const float*)d_in[12];
  const float* dWhh0   = (const float*)d_in[13];
  const float* dbih0   = (const float*)d_in[14];
  const float* dbhh0   = (const float*)d_in[15];
  const float* dWih1   = (const float*)d_in[16];
  const float* dWhh1   = (const float*)d_in[17];
  const float* dbih1   = (const float*)d_in[18];
  const float* dbhh1   = (const float*)d_in[19];
  const float* attn_W  = (const float*)d_in[20];
  const float* attn_b  = (const float*)d_in[21];
  const float* out_W   = (const float*)d_in[22];
  const float* out_b   = (const float*)d_in[23];
  float* out = (float*)d_out;

  // ---- workspace bump allocator ----
  char* p = (char*)d_ws;
  auto alloc  = [&](size_t nbytes) -> void* {
    void* r = (void*)p; p += (nbytes + 255) & ~(size_t)255; return r;
  };
  auto allocH = [&](size_t nelem) -> _Float16* { return (_Float16*)alloc(nelem * 2); };

  _Float16* xs      = allocH((size_t)S * B * E);
  _Float16* y0      = allocH((size_t)S * B * D);
  _Float16* y1      = allocH((size_t)S * B * D);
  _Float16* eWih0h  = allocH((size_t)2 * G3 * E);
  _Float16* eWhh0h  = allocH((size_t)2 * G3 * H);
  _Float16* eWih1h  = allocH((size_t)2 * G3 * D);
  _Float16* eWhh1h  = allocH((size_t)2 * G3 * H);
  _Float16* dWih0h  = allocH((size_t)DG3 * (E + D));
  _Float16* dWhh0h  = allocH((size_t)DG3 * D);
  _Float16* dWih1h  = allocH((size_t)DG3 * D);
  _Float16* dWhh1h  = allocH((size_t)DG3 * D);
  _Float16* outWh   = allocH((size_t)Vt * D);
  _Float16* tgtEmbH = allocH((size_t)Vt * E);
  _Float16* hz      = allocH((size_t)B * H);
  _Float16* hA      = allocH((size_t)B * H);
  _Float16* hB      = allocH((size_t)B * H);
  _Float16* dh0[2]  = { allocH((size_t)B * D), allocH((size_t)B * D) };
  _Float16* dh1[2]  = { allocH((size_t)B * D), allocH((size_t)B * D) };
  _Float16* xbuf    = allocH((size_t)B * (E + D));
  float* encScore   = (float*)alloc((size_t)B * S * 4);

  auto cvt = [&](const float* s_, _Float16* d_, size_t n) {
    f32_to_f16_k<<<dim3((unsigned)((n + 255) / 256)), dim3(256), 0, stream>>>(s_, d_, n);
  };

  // ---- weight / embedding conversion to f16 ----
  cvt(eWih0, eWih0h, (size_t)2 * G3 * E);
  cvt(eWhh0, eWhh0h, (size_t)2 * G3 * H);
  cvt(eWih1, eWih1h, (size_t)2 * G3 * D);
  cvt(eWhh1, eWhh1h, (size_t)2 * G3 * H);
  cvt(dWih0, dWih0h, (size_t)DG3 * (E + D));
  cvt(dWhh0, dWhh0h, (size_t)DG3 * D);
  cvt(dWih1, dWih1h, (size_t)DG3 * D);
  cvt(dWhh1, dWhh1h, (size_t)DG3 * D);
  cvt(out_W, outWh,  (size_t)Vt * D);
  cvt(tgt_emb, tgtEmbH, (size_t)Vt * E);

  {
    long total = (long)S * B * E;
    embed_src_k<<<dim3((unsigned)((total + 255) / 256)), dim3(256), 0, stream>>>(
        src, src_emb, xs, B, S, E, total);
  }
  zero_f16_k<<<dim3((B * H + 255) / 256), dim3(256), 0, stream>>>(hz, B * H);
  zero_t0_k<<<dim3((B * Vt + 255) / 256), dim3(256), 0, stream>>>(out, B, T, Vt);

  // ---- encoder layer 0 (both directions) ----
  for (int dir = 0; dir < 2; ++dir) {
    _Float16* hcur = hz;
    for (int s = 0; s < S; ++s) {
      const int os = dir ? (S - 1 - s) : s;
      _Float16* hnew; int ldhn;
      if (s == S - 1) { hnew = dh0[0] + dir * H; ldhn = D; }   // final state -> h0e
      else            { hnew = (s & 1) ? hB : hA; ldhn = H; }
      gru_step_wmma<<<dim3(H / 16), dim3(128), 0, stream>>>(
          xs + (size_t)os * B * E, E, E,
          hcur, H, H,
          eWih0h + (size_t)dir * G3 * E,
          eWhh0h + (size_t)dir * G3 * H,
          ebih0 + dir * G3, ebhh0 + dir * G3,
          hnew, ldhn,
          y0 + (size_t)os * B * D + dir * H, D);
      hcur = hnew;
    }
  }

  // ---- encoder layer 1 (both directions) ----
  for (int dir = 0; dir < 2; ++dir) {
    _Float16* hcur = hz;
    for (int s = 0; s < S; ++s) {
      const int os = dir ? (S - 1 - s) : s;
      _Float16* hnew; int ldhn;
      if (s == S - 1) { hnew = dh1[0] + dir * H; ldhn = D; }   // final state -> h1e
      else            { hnew = (s & 1) ? hB : hA; ldhn = H; }
      gru_step_wmma<<<dim3(H / 16), dim3(128), 0, stream>>>(
          y0 + (size_t)os * B * D, D, D,
          hcur, H, H,
          eWih1h + (size_t)dir * G3 * D,
          eWhh1h + (size_t)dir * G3 * H,
          ebih1 + dir * G3, ebhh1 + dir * G3,
          hnew, ldhn,
          y1 + (size_t)os * B * D + dir * H, D);
      hcur = hnew;
    }
  }

  // ---- precompute enc_score = enc_out @ We + attn_b ----
  enc_score_k<<<dim3((B * S) / 8), dim3(256), 0, stream>>>(
      y1, attn_W, attn_b, encScore, B, S, D);

  // ---- decoder loop ----
  int c0 = 0, c1 = 0;
  for (int t = 0; t < T - 1; ++t) {
    attn_ctx_k<<<dim3(B), dim3(256), 0, stream>>>(
        encScore, dh1[c1], D, attn_W, y1, tgt, t, T, tgtEmbH, xbuf, B, S, D, E);

    gru_step_wmma<<<dim3(D / 16), dim3(128), 0, stream>>>(
        xbuf, E + D, E + D,
        dh0[c0], D, D,
        dWih0h, dWhh0h, dbih0, dbhh0,
        dh0[c0 ^ 1], D,
        (_Float16*)nullptr, 0);

    gru_step_wmma<<<dim3(D / 16), dim3(128), 0, stream>>>(
        dh0[c0 ^ 1], D, D,
        dh1[c1], D, D,
        dWih1h, dWhh1h, dbih1, dbhh1,
        dh1[c1 ^ 1], D,
        (_Float16*)nullptr, 0);

    gemm_nt_wmma<<<dim3(Vt / 16), dim3(128), 0, stream>>>(
        dh1[c1 ^ 1], D, outWh, D, out_b,
        out + (size_t)(t + 1) * Vt, T * Vt, D);

    c0 ^= 1; c1 ^= 1;
  }
}
